// SolvGNNV2_37778532335670
// MI455X (gfx1250) — matrix operand order repaired and checked by reading, not compiled, not run on hardware
//
#include <hip/hip_runtime.h>
#include <hip/hip_bf16.h>

typedef __attribute__((ext_vector_type(16))) _Float16 v16h;
typedef __attribute__((ext_vector_type(8)))  float    v8f;

#define N_NODES  50000
#define N_EDGES  800000
#define N_GRAPHS 512
#define IN_DIM   74
#define HID      256

// ---------------------------------------------------------------- utilities

__global__ void zero_f32(float* __restrict__ p, int n) {
  int i = blockIdx.x * blockDim.x + threadIdx.x;
  for (; i < n; i += gridDim.x * blockDim.x) p[i] = 0.0f;
}

__device__ inline void atomAddF(float* p, float v) {
  unsafeAtomicAdd(p, v);   // -> global_atomic_add_f32 (L2 atomic unit)
}

// one thread per edge: accumulate out/in degrees
__global__ void edge_degrees(const int* __restrict__ src, const int* __restrict__ dst,
                             float* __restrict__ dout, float* __restrict__ din, int nE) {
  int e = blockIdx.x * blockDim.x + threadIdx.x;
  if (e >= nE) return;
  atomAddF(&dout[src[e]], 1.0f);
  atomAddF(&din[dst[e]], 1.0f);
}

__global__ void node_counts(const int* __restrict__ gid, float* __restrict__ cnt, int nV) {
  int v = blockIdx.x * blockDim.x + threadIdx.x;
  if (v >= nV) return;
  atomAddF(&cnt[gid[v]], 1.0f);
}

__global__ void make_scales(const float* __restrict__ deg, float* __restrict__ sc, int n) {
  int i = blockIdx.x * blockDim.x + threadIdx.x;
  if (i >= n) return;
  sc[i] = rsqrtf(fmaxf(deg[i], 1.0f));
}

// ---------------------------------------------------------------- SpMM
// one wave32 per edge; lanes stride the feature dim; src-side norm folded in
__global__ void spmm_kernel(const float* __restrict__ H, const float* __restrict__ sscale,
                            const int* __restrict__ src, const int* __restrict__ dst,
                            float* __restrict__ out, int dim, int nE) {
  int wave = blockIdx.x * (blockDim.x >> 5) + (threadIdx.x >> 5);
  if (wave >= nE) return;
  int lane = threadIdx.x & 31;
  int s = src[wave];
  int t = dst[wave];
  float sc = sscale[s];
  const float* __restrict__ hs = H + (size_t)s * dim;
  float* __restrict__ ot = out + (size_t)t * dim;
  for (int d = lane; d < dim; d += 32)
    atomAddF(&ot[d], hs[d] * sc);
}

// ---------------------------------------------------------------- weights -> f16, transposed + K-padded
// Bsw[n * Kpad + k] = (k < K) ? W[k * N + n] : 0   (column-major, zero pad)
__global__ void convert_weights(const float* __restrict__ W, _Float16* __restrict__ Bsw,
                                int K, int Kpad, int N) {
  int idx = blockIdx.x * blockDim.x + threadIdx.x;
  int total = Kpad * N;
  if (idx >= total) return;
  int n = idx / Kpad;
  int k = idx - n * Kpad;
  float v = (k < K) ? W[(size_t)k * N + n] : 0.0f;
  Bsw[(size_t)n * Kpad + k] = (_Float16)v;
}

// ---------------------------------------------------------------- WMMA GEMM
// C[M,N] = act( (rowScale .* A[M,K]) @ W[K,N] + bias )   (A f32, W pre-converted f16)
// block = 512 threads = 16 waves; block covers 16 rows x 256 cols; grid.y tiles N/256.
// ACT: 0 = none, 1 = relu, 2 = leaky(0.01)
template <int ACT>
__global__ void __launch_bounds__(512)
gemm_wmma(const float* __restrict__ A, const float* __restrict__ rowScale,
          const _Float16* __restrict__ Bsw, const float* __restrict__ bias,
          float* __restrict__ C, int M, int K, int Kpad, int N) {
  __shared__ _Float16 As[16 * 1024];          // up to Kpad = 1024 (32 KB)

  const int tid = threadIdx.x;
  const int r0  = blockIdx.x * 16;

  // ---- stage A rows (scaled, f16, zero-padded) into LDS
  const int total = 16 * Kpad;
  for (int idx = tid; idx < total; idx += 512) {
    int row = idx / Kpad;
    int k   = idx - row * Kpad;
    int gr  = r0 + row;
    float v = 0.0f;
    if (gr < M && k < K) {
      v = A[(size_t)gr * K + k];
      if (rowScale) v *= rowScale[gr];
    }
    As[row * Kpad + k] = (_Float16)v;
  }
  __syncthreads();

  const unsigned* As32 = (const unsigned*)As;  // packed f16 pairs
  const int lane  = tid & 31;
  const int wave  = tid >> 5;
  const int g     = lane >> 4;                 // half-wave group
  const int row16 = lane & 15;
  const int col   = blockIdx.y * 256 + wave * 16 + row16;   // this lane's output column
  const int khalf = Kpad >> 1;

  const _Float16* __restrict__ bbase = Bsw + (size_t)col * Kpad;

  v8f acc = {};
  const int nCh = Kpad >> 5;
  for (int c = 0; c < nCh; ++c) {
    const int kc = c << 5;
    // A fragment: 16x32 f16 layout; VGPR i holds K = 2*i (+1 hi) + 16*g
    union { v16h h; unsigned u[8]; } af;
#pragma unroll
    for (int i = 0; i < 8; ++i) {
      int kb = kc + (i << 1) + (g << 4);
      af.u[i] = As32[row16 * khalf + (kb >> 1)];
    }
    // B fragment: lanes 0-15 hold K=kc..kc+15, lanes 16-31 hold K=kc+16..kc+31 (contiguous 32B)
    v16h b = *(const v16h*)(bbase + kc + (g << 4));
    acc = __builtin_amdgcn_wmma_f32_16x16x32_f16(false, af.h, false, b,
                                                 (short)0, acc, false, false);
  }

  // ---- epilogue: C/D layout -> lane = col, VGPR v = row v + 8*g
  const float bv = bias[col];
#pragma unroll
  for (int v = 0; v < 8; ++v) {
    int grow = r0 + v + 8 * g;
    if (grow < M) {
      float o = acc[v] + bv;
      if (ACT == 1)      o = fmaxf(o, 0.0f);
      else if (ACT == 2) o = (o > 0.0f) ? o : 0.01f * o;
      C[(size_t)grow * N + col] = o;
    }
  }
}

// ---------------------------------------------------------------- pooling
__global__ void pool_sum(const float* __restrict__ H, const int* __restrict__ gid,
                         float* __restrict__ pool, int nV) {
  int idx = blockIdx.x * blockDim.x + threadIdx.x;
  if (idx >= nV * HID) return;
  int v = idx >> 8;           // HID == 256
  int d = idx & 255;
  atomAddF(&pool[(size_t)gid[v] * HID + d], H[(size_t)v * HID + d]);
}

__global__ void pool_div(float* __restrict__ pool, const float* __restrict__ cnt) {
  int idx = blockIdx.x * blockDim.x + threadIdx.x;
  if (idx >= N_GRAPHS * HID) return;
  int gidx = idx >> 8;
  pool[idx] *= (1.0f / fmaxf(cnt[gidx], 1.0f));
}

// ---------------------------------------------------------------- final 512-dot head
__global__ void head_out(const float* __restrict__ z, const float* __restrict__ w,
                         const float* __restrict__ b, float* __restrict__ out) {
  int gidx = blockIdx.x * (blockDim.x >> 5) + (threadIdx.x >> 5);
  int lane = threadIdx.x & 31;
  if (gidx >= N_GRAPHS) return;
  const float* __restrict__ zr = z + (size_t)gidx * 512;
  float s = 0.0f;
  for (int k = lane; k < 512; k += 32) s += zr[k] * w[k];
#pragma unroll
  for (int off = 16; off; off >>= 1) s += __shfl_xor(s, off, 32);
  if (lane == 0) out[gidx] = s + b[0];
}

// ---------------------------------------------------------------- launch

extern "C" void kernel_launch(void* const* d_in, const int* in_sizes, int n_in,
                              void* d_out, int out_size, void* d_ws, size_t ws_size,
                              hipStream_t stream) {
  const float* x    = (const float*)d_in[0];
  const int*   src  = (const int*)  d_in[1];
  const int*   dst  = (const int*)  d_in[2];
  const int*   gid  = (const int*)  d_in[3];
  const float* W1   = (const float*)d_in[4];
  const float* b1   = (const float*)d_in[5];
  const float* W2   = (const float*)d_in[6];
  const float* b2   = (const float*)d_in[7];
  const float* Wm1  = (const float*)d_in[8];
  const float* bm1  = (const float*)d_in[9];
  const float* Wm2  = (const float*)d_in[10];
  const float* bm2  = (const float*)d_in[11];
  const float* Wm3  = (const float*)d_in[12];
  const float* bm3  = (const float*)d_in[13];
  float* out = (float*)d_out;

  // ---- workspace carve-out (256 B aligned)
  char* base = (char*)d_ws;
  size_t off = 0;
  auto carve = [&](size_t bytes) -> char* {
    size_t start = (off + 255) & ~(size_t)255;
    off = start + bytes;
    return base + start;
  };
  float*    deg_out  = (float*)   carve(N_NODES * 4);
  float*    deg_in   = (float*)   carve(N_NODES * 4);
  float*    so_scale = (float*)   carve(N_NODES * 4);
  float*    si_scale = (float*)   carve(N_NODES * 4);
  float*    cnt      = (float*)   carve(N_GRAPHS * 4);
  float*    m1       = (float*)   carve((size_t)N_NODES * IN_DIM * 4);
  float*    h1       = (float*)   carve((size_t)N_NODES * HID * 4);
  float*    m2       = (float*)   carve((size_t)N_NODES * HID * 4);  // h2 in-place
  float*    pool     = (float*)   carve((size_t)N_GRAPHS * HID * 4);
  float*    z1       = (float*)   carve((size_t)N_GRAPHS * 1024 * 4);
  float*    z2       = (float*)   carve((size_t)N_GRAPHS * 512 * 4);
  _Float16* W1h      = (_Float16*)carve((size_t)96   * 256  * 2);
  _Float16* W2h      = (_Float16*)carve((size_t)256  * 256  * 2);
  _Float16* Wm1h     = (_Float16*)carve((size_t)256  * 1024 * 2);
  _Float16* Wm2h     = (_Float16*)carve((size_t)1024 * 512  * 2);
  (void)ws_size; (void)n_in; (void)in_sizes; (void)out_size;

  const int T = 256;
  auto cdiv = [](long long a, long long b) { return (int)((a + b - 1) / b); };

  // ---- zero accumulators
  zero_f32<<<cdiv(N_NODES, T), T, 0, stream>>>(deg_out, N_NODES);
  zero_f32<<<cdiv(N_NODES, T), T, 0, stream>>>(deg_in,  N_NODES);
  zero_f32<<<cdiv(N_GRAPHS, T), T, 0, stream>>>(cnt, N_GRAPHS);
  zero_f32<<<cdiv((long long)N_NODES * IN_DIM, T), T, 0, stream>>>(m1, N_NODES * IN_DIM);
  zero_f32<<<cdiv((long long)N_NODES * HID, T), T, 0, stream>>>(m2, N_NODES * HID);
  zero_f32<<<cdiv((long long)N_GRAPHS * HID, T), T, 0, stream>>>(pool, N_GRAPHS * HID);

  // ---- degrees, counts, norms
  edge_degrees<<<cdiv(N_EDGES, T), T, 0, stream>>>(src, dst, deg_out, deg_in, N_EDGES);
  node_counts<<<cdiv(N_NODES, T), T, 0, stream>>>(gid, cnt, N_NODES);
  make_scales<<<cdiv(N_NODES, T), T, 0, stream>>>(deg_out, so_scale, N_NODES);
  make_scales<<<cdiv(N_NODES, T), T, 0, stream>>>(deg_in,  si_scale, N_NODES);

  // ---- weights -> f16 column-major (K padded to multiple of 32)
  convert_weights<<<cdiv(96 * 256, T), T, 0, stream>>>(W1, W1h, IN_DIM, 96, 256);
  convert_weights<<<cdiv(256 * 256, T), T, 0, stream>>>(W2, W2h, 256, 256, 256);
  convert_weights<<<cdiv(256 * 1024, T), T, 0, stream>>>(Wm1, Wm1h, 256, 256, 1024);
  convert_weights<<<cdiv(1024 * 512, T), T, 0, stream>>>(Wm2, Wm2h, 1024, 1024, 512);

  const int wavesPerBlk = T >> 5;                     // 8 edges per block
  const int spmmBlks = cdiv(N_EDGES, wavesPerBlk);

  // ---- layer 1: aggregate then GEMM(74 -> 256), relu
  spmm_kernel<<<spmmBlks, T, 0, stream>>>(x, so_scale, src, dst, m1, IN_DIM, N_EDGES);
  gemm_wmma<1><<<dim3(cdiv(N_NODES, 16), 1), 512, 0, stream>>>(
      m1, si_scale, W1h, b1, h1, N_NODES, IN_DIM, 96, 256);

  // ---- layer 2: aggregate then GEMM(256 -> 256), relu (in-place m2 -> h2)
  spmm_kernel<<<spmmBlks, T, 0, stream>>>(h1, so_scale, src, dst, m2, HID, N_EDGES);
  gemm_wmma<1><<<dim3(cdiv(N_NODES, 16), 1), 512, 0, stream>>>(
      m2, si_scale, W2h, b2, m2, N_NODES, 256, 256, 256);

  // ---- per-graph mean pooling
  pool_sum<<<cdiv((long long)N_NODES * HID, T), T, 0, stream>>>(m2, gid, pool, N_NODES);
  pool_div<<<cdiv(N_GRAPHS * HID, T), T, 0, stream>>>(pool, cnt);

  // ---- MLP head: 256 -> 1024 -> 512 -> 1 (leaky relu 0.01)
  gemm_wmma<2><<<dim3(cdiv(N_GRAPHS, 16), 4), 512, 0, stream>>>(
      pool, nullptr, Wm1h, bm1, z1, N_GRAPHS, 256, 256, 1024);
  gemm_wmma<2><<<dim3(cdiv(N_GRAPHS, 16), 2), 512, 0, stream>>>(
      z1, nullptr, Wm2h, bm2, z2, N_GRAPHS, 1024, 1024, 512);
  head_out<<<cdiv(N_GRAPHS, wavesPerBlk), T, 0, stream>>>(z2, Wm3, bm3, out);
}
